// HeteroGAT_63651415327105
// MI455X (gfx1250) — matrix operand order repaired and checked by reading, not compiled, not run on hardware
//
#include <hip/hip_runtime.h>
#include <math.h>

// ---------------------------------------------------------------------------
// HeteroGAT forward for MI455X (gfx1250, wave32).
//  - All dense projections: v_wmma_f32_16x16x32_bf16, operands pre-converted
//    to bf16 so fragments are pure global_load_b128 (no per-element VALU).
//  - Weights pre-transposed to [N][K] bf16 so B fragments are contiguous.
//  - Edge softmax/aggregation: native u32/f32 global atomics (L2-resident).
// ---------------------------------------------------------------------------

#define NEG_SLOPE 0.2f
#define LN_EPS 1e-5f

typedef __attribute__((ext_vector_type(16))) __bf16   v16bf;
typedef __attribute__((ext_vector_type(8)))  float    v8f;
typedef __attribute__((ext_vector_type(4)))  unsigned uint4v;

union Frag { uint4v q[2]; v16bf h; };      // 32 bytes = one WMMA A/B operand

static __device__ __forceinline__ unsigned short f2bf_u16(float f) {
  unsigned u = __float_as_uint(f);
  unsigned r = (u + 0x7fffu + ((u >> 16) & 1u)) >> 16;   // round-to-nearest-even
  return (unsigned short)r;
}

// ---------------------------------------------------------------------------
// packed f32 -> bf16 conversion, 2 elements per thread (dword stores)
// ---------------------------------------------------------------------------
__global__ void k_cvt_bf16(const float* __restrict__ in, unsigned* __restrict__ out, long n2)
{
  long i = (long)blockIdx.x * blockDim.x + threadIdx.x;
  if (i >= n2) return;
  unsigned lo = f2bf_u16(in[2 * i]);
  unsigned hi = f2bf_u16(in[2 * i + 1]);
  out[i] = lo | (hi << 16);
}

// W[K,N] f32 -> Wt[N,K] bf16 (weights are tiny; runs once per GEMM)
__global__ void k_cvt_bf16_t(const float* __restrict__ W, unsigned short* __restrict__ Wt,
                             int K, int N)
{
  int t = blockIdx.x * blockDim.x + threadIdx.x;
  if (t >= K * N) return;
  int n = t / K, k = t - n * K;
  Wt[t] = f2bf_u16(W[(size_t)k * N + n]);
}

// ---------------------------------------------------------------------------
// C[M,N] = A[M,K](bf16) @ Wt[N,K](bf16)^T (+ bias[N])
// One wave computes a 16x64 strip: 4 accumulators share one A fragment.
// M % 16 == 0, N % 64 == 0, K % 32 == 0 (holds for every GEMM here).
// ---------------------------------------------------------------------------
__global__ void k_wmma_gemm4(const unsigned short* __restrict__ A,
                             const unsigned short* __restrict__ Wt,
                             const float* __restrict__ bias, float* __restrict__ C,
                             int M, int K, int N)
{
  const int lane = threadIdx.x & 31;
  const int wave = threadIdx.x >> 5;
  const int nstrips = N >> 6;
  const int strip = blockIdx.x * (blockDim.x >> 5) + wave;
  const int total = (M >> 4) * nstrips;
  if (strip >= total) return;                // wave-uniform: EXEC stays all-ones
  const int mt = strip / nstrips;
  const int n0 = (strip - mt * nstrips) << 6;

  const int row  = lane & 15;                // A: one row per lane
  const int kgrp = lane >> 4;                // lane half selects K sub-block
  const int colB = lane & 15;                // B/C/D: one column per lane

  v8f acc0 = {}, acc1 = {}, acc2 = {}, acc3 = {};
  const unsigned short* arow = A + (size_t)(mt * 16 + row) * K;
  const unsigned short* b0 = Wt + (size_t)(n0 +  0 + colB) * K;
  const unsigned short* b1 = Wt + (size_t)(n0 + 16 + colB) * K;
  const unsigned short* b2 = Wt + (size_t)(n0 + 32 + colB) * K;
  const unsigned short* b3 = Wt + (size_t)(n0 + 48 + colB) * K;

  for (int k0 = 0; k0 < K; k0 += 32) {
    if (k0 + 32 < K) __builtin_prefetch(arow + k0 + 32, 0, 0);  // global_prefetch_b8
    // A fragment: 16 contiguous halves at K = k0 + kgrp*16 (matches B)
    Frag fa;
    const uint4v* pa = (const uint4v*)(arow + k0 + kgrp * 16);
    fa.q[0] = pa[0];
    fa.q[1] = pa[1];
    // B fragments: 16 contiguous halves at K = k0 + kgrp*16
    const size_t bk = (size_t)k0 + (size_t)kgrp * 16;
    Frag fb0, fb1, fb2, fb3;
    { const uint4v* p = (const uint4v*)(b0 + bk); fb0.q[0] = p[0]; fb0.q[1] = p[1]; }
    { const uint4v* p = (const uint4v*)(b1 + bk); fb1.q[0] = p[0]; fb1.q[1] = p[1]; }
    { const uint4v* p = (const uint4v*)(b2 + bk); fb2.q[0] = p[0]; fb2.q[1] = p[1]; }
    { const uint4v* p = (const uint4v*)(b3 + bk); fb3.q[0] = p[0]; fb3.q[1] = p[1]; }

    acc0 = __builtin_amdgcn_wmma_f32_16x16x32_bf16(false, fa.h, false, fb0.h, (short)0, acc0, false, false);
    acc1 = __builtin_amdgcn_wmma_f32_16x16x32_bf16(false, fa.h, false, fb1.h, (short)0, acc1, false, false);
    acc2 = __builtin_amdgcn_wmma_f32_16x16x32_bf16(false, fa.h, false, fb2.h, (short)0, acc2, false, false);
    acc3 = __builtin_amdgcn_wmma_f32_16x16x32_bf16(false, fa.h, false, fb3.h, (short)0, acc3, false, false);
  }

  // C/D layout: element r -> row = r + 8*(lane>=16), col = lane&15
  const int rbase = mt * 16 + kgrp * 8;
#pragma unroll
  for (int j = 0; j < 4; ++j) {
    const v8f acc = (j == 0) ? acc0 : (j == 1) ? acc1 : (j == 2) ? acc2 : acc3;
    const int cc = n0 + j * 16 + colB;
    const float bv = bias ? bias[cc] : 0.0f;
#pragma unroll
    for (int r = 0; r < 8; ++r)
      C[(size_t)(rbase + r) * N + cc] = acc[r] + bv;
  }
}

// w[k*H+h] = sum_d fc[k, h*D+d] * attn[h,d]   (folds fd projection into a matvec)
__global__ void k_make_w(const float* __restrict__ fc, const float* __restrict__ attn,
                         float* __restrict__ w, int K, int H, int D)
{
  int t = blockIdx.x * blockDim.x + threadIdx.x;
  if (t >= K * H) return;
  int k = t / H, h = t - k * H;
  const float* fr = fc + (size_t)k * H * D + h * D;
  const float* ar = attn + h * D;
  float s = 0.f;
  for (int d = 0; d < D; ++d) s += fr[d] * ar[d];
  w[t] = s;
}

// out[n*H+h] = sum_k X[n,k] * w[k*H+h]
__global__ void k_matvec(const float* __restrict__ X, const float* __restrict__ w,
                         float* __restrict__ out, int Nn, int K, int H)
{
  int t = blockIdx.x * blockDim.x + threadIdx.x;
  if (t >= Nn * H) return;
  int n = t / H, h = t - n * H;
  const float* xr = X + (size_t)n * K;
  float s = 0.f;
  for (int k = 0; k < K; ++k) s += xr[k] * w[k * H + h];
  out[t] = s;
}

__global__ void k_fill_zero(float* __restrict__ p, long n)
{
  long i = (long)blockIdx.x * blockDim.x + threadIdx.x;
  if (i < n) p[i] = 0.0f;
}

__global__ void k_fill_bias(float* __restrict__ out, const float* __restrict__ bias,
                            long total, int F)
{
  long i = (long)blockIdx.x * blockDim.x + threadIdx.x;
  if (i < total) out[i] = bias[i % F];
}

// pass 1: e = leaky_relu(el[src]+er[dst]); segment max via monotone uint atomicMax
__global__ void k_edge1(const int* __restrict__ src, const int* __restrict__ dst,
                        const float* __restrict__ el, const float* __restrict__ er,
                        float* __restrict__ ebuf, unsigned* __restrict__ mbuf,
                        int E, int H)
{
  long idx = (long)blockIdx.x * blockDim.x + threadIdx.x;
  if (idx >= (long)E * H) return;
  int e = (int)(idx / H), h = (int)(idx - (long)e * H);
  int s = src[e], d = dst[e];
  float x = el[s * H + h] + er[d * H + h];
  x = x > 0.f ? x : NEG_SLOPE * x;
  ebuf[idx] = x;
  unsigned u = __float_as_uint(x);
  u = (u & 0x80000000u) ? ~u : (u | 0x80000000u);     // order-preserving map
  __hip_atomic_fetch_max(&mbuf[d * H + h], u, __ATOMIC_RELAXED, __HIP_MEMORY_SCOPE_AGENT);
}

// pass 2: ex = exp(e - m[dst]); segment sum
__global__ void k_edge2(const int* __restrict__ dst, float* __restrict__ ebuf,
                        const unsigned* __restrict__ mbuf, float* __restrict__ sbuf,
                        int E, int H)
{
  long idx = (long)blockIdx.x * blockDim.x + threadIdx.x;
  if (idx >= (long)E * H) return;
  int e = (int)(idx / H), h = (int)(idx - (long)e * H);
  int d = dst[e];
  unsigned u = mbuf[d * H + h];
  unsigned b = (u & 0x80000000u) ? (u & 0x7fffffffu) : ~u;
  float ex = __expf(ebuf[idx] - __uint_as_float(b));
  ebuf[idx] = ex;
  __hip_atomic_fetch_add(&sbuf[d * H + h], ex, __ATOMIC_RELAXED, __HIP_MEMORY_SCOPE_AGENT);
}

// pass 3: out[dst, h, :] += alpha * fs[src, h, :]   (one wave per (edge, head))
__global__ void k_edge3(const int* __restrict__ src, const int* __restrict__ dst,
                        const float* __restrict__ ebuf, const float* __restrict__ sbuf,
                        const float* __restrict__ fs, float* __restrict__ out,
                        int E, int H, int D)
{
  int lane = threadIdx.x & 31;
  long w = (long)blockIdx.x * (blockDim.x >> 5) + (threadIdx.x >> 5);
  if (w >= (long)E * H) return;              // wave-uniform
  int e = (int)(w / H), h = (int)(w - (long)e * H);
  int s = src[e], d = dst[e];
  float alpha = ebuf[w] / fmaxf(sbuf[d * H + h], 1e-9f);
  int F = H * D;
  const float* fsrc = fs + (size_t)s * F + h * D;
  float* od = out + (size_t)d * F + h * D;
  for (int t = lane; t < D; t += 32)
    __hip_atomic_fetch_add(&od[t], alpha * fsrc[t], __ATOMIC_RELAXED, __HIP_MEMORY_SCOPE_AGENT);
}

// layernorm + elu, in place; one wave per node, F = 256
__global__ void k_ln_elu(float* __restrict__ x, const float* __restrict__ g,
                         const float* __restrict__ b, int Nn, int F)
{
  int lane = threadIdx.x & 31;
  int node = blockIdx.x * (blockDim.x >> 5) + (threadIdx.x >> 5);
  if (node >= Nn) return;
  float* row = x + (size_t)node * F;
  const int per = F >> 5;                    // 8
  float v[8];
  float s = 0.f;
  for (int i = 0; i < per; ++i) { v[i] = row[lane + i * 32]; s += v[i]; }
  for (int m = 16; m >= 1; m >>= 1) s += __shfl_xor(s, m, 32);
  float mu = s / (float)F;
  float var = 0.f;
  for (int i = 0; i < per; ++i) { float d = v[i] - mu; var += d * d; }
  for (int m = 16; m >= 1; m >>= 1) var += __shfl_xor(var, m, 32);
  float inv = rsqrtf(var / (float)F + LN_EPS);
  for (int i = 0; i < per; ++i) {
    int f = lane + i * 32;
    float y = (v[i] - mu) * inv * g[f] + b[f];
    row[f] = y > 0.f ? y : (__expf(y) - 1.f);
  }
}

// ---------------------------------------------------------------------------
extern "C" void kernel_launch(void* const* d_in, const int* in_sizes, int n_in,
                              void* d_out, int out_size, void* d_ws, size_t ws_size,
                              hipStream_t stream)
{
  (void)in_sizes; (void)n_in; (void)out_size; (void)ws_size;
  const int NU = 20000, NI = 20000, E = 320000;
  const int DIN = 256, HID = 64, H = 4, OUT = 64;

  // ---- inputs, flattened in setup_inputs() dict (insertion) order ----
  const float* x_user = (const float*)d_in[0];
  const float* x_item = (const float*)d_in[1];
  const int*   src_uc = (const int*)d_in[2];
  const int*   dst_uc = (const int*)d_in[3];
  const int*   src_cb = (const int*)d_in[4];
  const int*   dst_cb = (const int*)d_in[5];
  const float* pu_w = (const float*)d_in[6];
  const float* pu_b = (const float*)d_in[7];
  const float* pi_w = (const float*)d_in[8];
  const float* pi_b = (const float*)d_in[9];
  const float* l0uc_fc = (const float*)d_in[10];
  const float* l0uc_al = (const float*)d_in[11];
  const float* l0uc_ar = (const float*)d_in[12];
  const float* l0uc_b  = (const float*)d_in[13];
  const float* l0uc_rs = (const float*)d_in[14];
  const float* l0cb_fc = (const float*)d_in[15];
  const float* l0cb_al = (const float*)d_in[16];
  const float* l0cb_ar = (const float*)d_in[17];
  const float* l0cb_b  = (const float*)d_in[18];
  const float* l0cb_rs = (const float*)d_in[19];
  const float* l1uc_fc = (const float*)d_in[20];
  const float* l1uc_al = (const float*)d_in[21];
  const float* l1uc_ar = (const float*)d_in[22];
  const float* l1uc_b  = (const float*)d_in[23];
  const float* l1cb_fc = (const float*)d_in[24];
  const float* l1cb_al = (const float*)d_in[25];
  const float* l1cb_ar = (const float*)d_in[26];
  const float* l1cb_b  = (const float*)d_in[27];
  const float* ln_g = (const float*)d_in[28];
  const float* ln_b = (const float*)d_in[29];

  // ---- workspace carve-out (float units; all chunks 16B-aligned) ----
  float* ws = (float*)d_ws;
  size_t off = 0;
  auto carve = [&](size_t nf) { float* p = ws + off; off += nf; return p; };
  float* hu  = carve((size_t)NU * HID);
  float* hi  = carve((size_t)NI * HID);
  float* h1u = carve((size_t)NU * H * HID);
  float* h1i = carve((size_t)NI * H * HID);
  float* fsb = carve((size_t)NU * H * HID);     // src projection, reused per conv
  float* el  = carve((size_t)NU * H);
  float* er  = carve((size_t)NU * H);
  float* wl  = carve(1024);
  float* wr  = carve(1024);
  float* mb  = carve((size_t)NU * H);           // reinterpreted as unsigned
  float* sb  = carve((size_t)NU * H);
  float* eb  = carve((size_t)E * H);
  // bf16 operand buffers (counts in halves -> /2 floats)
  unsigned short* xu_b  = (unsigned short*)carve((size_t)NU * DIN / 2);
  unsigned short* xi_b  = (unsigned short*)carve((size_t)NI * DIN / 2);
  unsigned short* hu_b  = (unsigned short*)carve((size_t)NU * HID / 2);
  unsigned short* hi_b  = (unsigned short*)carve((size_t)NI * HID / 2);
  unsigned short* h1u_b = (unsigned short*)carve((size_t)NU * H * HID / 2);
  unsigned short* h1i_b = (unsigned short*)carve((size_t)NI * H * HID / 2);
  unsigned short* wt    = (unsigned short*)carve(32768);   // transposed bf16 weight, reused

  auto cvt = [&](const float* src, unsigned short* dst, long n) {
    long n2 = n / 2;
    k_cvt_bf16<<<(int)((n2 + 255) / 256), 256, 0, stream>>>(src, (unsigned*)dst, n2);
  };
  // C = A(bf16)[M,K] @ W(f32)[K,N] + bias : transpose-convert W, then WMMA
  auto gemm = [&](const unsigned short* A, const float* W, const float* bias, float* C,
                  int M, int K, int N) {
    k_cvt_bf16_t<<<(K * N + 255) / 256, 256, 0, stream>>>(W, wt, K, N);
    int strips = (M / 16) * (N / 64);
    k_wmma_gemm4<<<(strips + 7) / 8, 256, 0, stream>>>(A, wt, bias, C, M, K, N);
  };

  // one GAT conv: src feats (f32 + bf16 views) -> out[Ndst, heads*dout]
  auto gat = [&](const float* hsrc, const unsigned short* hsrc_b, const float* hdst,
                 const unsigned short* hdst_b, const int* srcI, const int* dstI,
                 const float* fc, const float* al, const float* ar,
                 const float* bias, const float* res, float* out,
                 int Nsrc, int Ndst, int Kin, int heads, int dout) {
    int F = heads * dout;
    k_make_w<<<(Kin * heads + 255) / 256, 256, 0, stream>>>(fc, al, wl, Kin, heads, dout);
    k_make_w<<<(Kin * heads + 255) / 256, 256, 0, stream>>>(fc, ar, wr, Kin, heads, dout);
    k_matvec<<<(Nsrc * heads + 255) / 256, 256, 0, stream>>>(hsrc, wl, el, Nsrc, Kin, heads);
    k_matvec<<<(Ndst * heads + 255) / 256, 256, 0, stream>>>(hdst, wr, er, Ndst, Kin, heads);
    gemm(hsrc_b, fc, nullptr, fsb, Nsrc, Kin, F);                 // fs = hsrc @ fc
    if (res) gemm(hdst_b, res, bias, out, Ndst, Kin, F);          // out = hdst@res + bias
    else k_fill_bias<<<(int)(((long)Ndst * F + 255) / 256), 256, 0, stream>>>(
             out, bias, (long)Ndst * F, F);
    k_fill_zero<<<(Ndst * heads + 255) / 256, 256, 0, stream>>>(mb, (long)Ndst * heads);
    k_fill_zero<<<(Ndst * heads + 255) / 256, 256, 0, stream>>>(sb, (long)Ndst * heads);
    long eh = (long)E * heads;
    k_edge1<<<(int)((eh + 255) / 256), 256, 0, stream>>>(srcI, dstI, el, er, eb,
                                                         (unsigned*)mb, E, heads);
    k_edge2<<<(int)((eh + 255) / 256), 256, 0, stream>>>(dstI, eb, (const unsigned*)mb,
                                                         sb, E, heads);
    k_edge3<<<(int)((eh * 32 + 255) / 256), 256, 0, stream>>>(srcI, dstI, eb, sb, fsb,
                                                              out, E, heads, dout);
  };

  // ---- input projections ----
  cvt(x_user, xu_b, (long)NU * DIN);
  cvt(x_item, xi_b, (long)NI * DIN);
  gemm(xu_b, pu_w, pu_b, hu, NU, DIN, HID);
  gemm(xi_b, pi_w, pi_b, hi, NI, DIN, HID);
  cvt(hu, hu_b, (long)NU * HID);
  cvt(hi, hi_b, (long)NI * HID);

  // ---- layer 0 (4 heads, residual): uc -> items, cb -> users ----
  gat(hu, hu_b, hi, hi_b, src_uc, dst_uc, l0uc_fc, l0uc_al, l0uc_ar, l0uc_b, l0uc_rs,
      h1i, NU, NI, HID, H, HID);
  gat(hi, hi_b, hu, hu_b, src_cb, dst_cb, l0cb_fc, l0cb_al, l0cb_ar, l0cb_b, l0cb_rs,
      h1u, NI, NU, HID, H, HID);

  // ---- layernorm + elu, then bf16 views for layer-1 GEMMs ----
  k_ln_elu<<<(NU + 7) / 8, 256, 0, stream>>>(h1u, ln_g, ln_b, NU, H * HID);
  k_ln_elu<<<(NI + 7) / 8, 256, 0, stream>>>(h1i, ln_g, ln_b, NI, H * HID);
  cvt(h1u, h1u_b, (long)NU * H * HID);
  cvt(h1i, h1i_b, (long)NI * H * HID);

  // ---- layer 1 (1 head, no residual); outputs (hu2, hi2) concatenated ----
  float* hu2 = (float*)d_out;
  float* hi2 = (float*)d_out + (size_t)NU * OUT;
  gat(h1u, h1u_b, h1i, h1i_b, src_uc, dst_uc, l1uc_fc, l1uc_al, l1uc_ar, l1uc_b, nullptr,
      hi2, NU, NI, H * HID, 1, OUT);
  gat(h1i, h1i_b, h1u, h1u_b, src_cb, dst_cb, l1cb_fc, l1cb_al, l1cb_ar, l1cb_b, nullptr,
      hu2, NI, NU, H * HID, 1, OUT);
}
